// TreeEncoderRNNNew_712964571852
// MI455X (gfx1250) — compile-verified, matrix operand order checked by reading, not run on hardware
//
#include <hip/hip_runtime.h>
#include <hip/hip_bf16.h>

// ---------------------------------------------------------------------------
// Types for CDNA5 WMMA (wave32)
// ---------------------------------------------------------------------------
typedef __bf16 bf16_t;
typedef bf16_t v16bf __attribute__((ext_vector_type(16)));
typedef bf16_t v8bf  __attribute__((ext_vector_type(8)));
typedef float  v8f   __attribute__((ext_vector_type(8)));
typedef float  v4f   __attribute__((ext_vector_type(4)));

#define H 256
#define W_ELEMS 65536               // 256*256 per weight matrix
#define L_TOKENS 65536
#define M_TILE 32                   // output rows per workgroup (2 x 16 subtiles)

__device__ __forceinline__ v8f wmma_bf16(v16bf a, v16bf b, v8f c) {
  // D = A(16x32 bf16) * B(32x16 bf16) + C(16x16 f32)
  return __builtin_amdgcn_wmma_f32_16x16x32_bf16(
      /*neg_a=*/false, a, /*neg_b=*/false, b,
      /*c_mod=*/(short)0, c, /*reuse_a=*/false, /*reuse_b=*/false);
}

// A-operand gather from an f32 row (row-major), converting to bf16.
// 16-bit A layout: this lane needs K = [kbase..kbase+7] in regs 0-3 and
// K = [kbase+16..kbase+23] in regs 4-7 (kbase includes the (lane>>4)*8 part).
__device__ __forceinline__ v16bf load_a_from_f32(const float* __restrict__ row,
                                                 int kbase) {
  v4f f0 = *reinterpret_cast<const v4f*>(row + kbase);
  v4f f1 = *reinterpret_cast<const v4f*>(row + kbase + 4);
  v4f f2 = *reinterpret_cast<const v4f*>(row + kbase + 16);
  v4f f3 = *reinterpret_cast<const v4f*>(row + kbase + 20);
  v16bf a;
#pragma unroll
  for (int i = 0; i < 4; ++i) {
    a[i]      = (bf16_t)f0[i];
    a[4 + i]  = (bf16_t)f1[i];
    a[8 + i]  = (bf16_t)f2[i];
    a[12 + i] = (bf16_t)f3[i];
  }
  return a;
}

// Loader for bf16 rows (B from global weights, A tiles staged in LDS):
// 8 bf16 at kbase, 8 bf16 at kbase+16.
__device__ __forceinline__ v16bf load_ab_bf16(const bf16_t* row, int kbase) {
  v8bf lo = *reinterpret_cast<const v8bf*>(row + kbase);
  v8bf hi = *reinterpret_cast<const v8bf*>(row + kbase + 16);
  v16bf r;
#pragma unroll
  for (int i = 0; i < 8; ++i) { r[i] = lo[i]; r[8 + i] = hi[i]; }
  return r;
}

__device__ __forceinline__ float fsigmoid(float x) {
  return 1.0f / (1.0f + __expf(-x));
}

// ---------------------------------------------------------------------------
// Kernel 1: pack all 12 weight matrices (f32) to bf16:
//   [WL(5) | WR(5) | Wlh | Wrh], each 256x256 row-major (o-major, h contig).
// ---------------------------------------------------------------------------
__global__ void cvt_weights_kernel(const float* __restrict__ WL,
                                   const float* __restrict__ WR,
                                   const float* __restrict__ Wlh,
                                   const float* __restrict__ Wrh,
                                   bf16_t* __restrict__ wb) {
  int i = blockIdx.x * blockDim.x + threadIdx.x;
  if (i >= 12 * W_ELEMS) return;
  float v;
  if (i < 5 * W_ELEMS)       v = WL[i];
  else if (i < 10 * W_ELEMS) v = WR[i - 5 * W_ELEMS];
  else if (i < 11 * W_ELEMS) v = Wlh[i - 10 * W_ELEMS];
  else                       v = Wrh[i - 11 * W_ELEMS];
  wb[i] = (bf16_t)v;
}

// ---------------------------------------------------------------------------
// Kernel 2: embedding gather  cur[i,:] = emb[tokens[i],:]
// ---------------------------------------------------------------------------
__global__ void gather_kernel(const int* __restrict__ tokens,
                              const float* __restrict__ emb,
                              float* __restrict__ cur) {
  int i = blockIdx.x;           // token index
  int t = threadIdx.x;          // 0..63, each thread moves a float4
  long long tok = tokens[i];
  const v4f* src = reinterpret_cast<const v4f*>(emb + tok * H);
  v4f* dst = reinterpret_cast<v4f*>(cur + (long long)i * H);
  dst[t] = src[t];
}

// ---------------------------------------------------------------------------
// Kernel 3: one tree level.  N = number of output rows (pairs).
// Workgroup = 256 threads (8 waves), handles 32 output rows (2 M-subtiles).
// Work partition: 16 N-chunk jobs, exactly 2 per wave; each job computes all
// 5 gates (B operands reused across both M-subtiles, A reused across gates).
// The same wave later runs the candidate GEMM for the same N-chunk, so the
// update-gate combine terms stay in registers across the barrier.
// ---------------------------------------------------------------------------
__global__ void __launch_bounds__(256)
tree_level_kernel(const float* __restrict__ curIn,   // 2N x 256
                  float* __restrict__ curOut,        // N  x 256
                  const bf16_t* __restrict__ wb,     // packed bf16 weights
                  const float* __restrict__ bL,      // 5 x 256
                  const float* __restrict__ bR,      // 5 x 256
                  const float* __restrict__ blh,     // 256
                  const float* __restrict__ brh,     // 256
                  int N) {
  __shared__ bf16_t uS[M_TILE * H];   // r_l * left  (bf16), 16 KB
  __shared__ bf16_t vS[M_TILE * H];   // r_r * right (bf16), 16 KB

  const int tid  = threadIdx.x;
  const int lane = tid & 31;
  const int wave = tid >> 5;
  const int lo   = lane & 15;     // A: row-within-subtile / B: output column
  const int hi   = lane >> 4;     // K sub-group select
  const int tileBase = blockIdx.x * M_TILE;

  // Clamped A-operand row pointers for the two M-subtiles.
  int m0 = tileBase + lo;       if (m0 > N - 1) m0 = N - 1;
  int m1 = tileBase + 16 + lo;  if (m1 > N - 1) m1 = N - 1;
  const float* l0 = curIn + (size_t)(2 * m0) * H;
  const float* r0 = l0 + H;
  const float* l1 = curIn + (size_t)(2 * m1) * H;
  const float* r1 = l1 + H;

  // Persisted across the barrier: part = z_l*left + z_r*right, and z.
  float part[2][2][8];
  float zzs [2][2][8];

  // ---------------- Phase 1: all 5 gates per N-chunk ----------------------
#pragma unroll
  for (int jj = 0; jj < 2; ++jj) {
    const int nbase = (wave + jj * 8) * 16;

    const bf16_t* wrow[10];
#pragma unroll
    for (int g = 0; g < 10; ++g)
      wrow[g] = wb + (size_t)g * W_ELEMS + (size_t)(nbase + lo) * H;
    __builtin_prefetch(wrow[0], 0, 1);

    v8f acc[5][2];
#pragma unroll
    for (int g = 0; g < 5; ++g)
#pragma unroll
      for (int st = 0; st < 2; ++st)
        acc[g][st] = (v8f){0.f, 0.f, 0.f, 0.f, 0.f, 0.f, 0.f, 0.f};

#pragma unroll
    for (int kc = 0; kc < 8; ++kc) {
      const int kbase = kc * 32 + hi * 8;
      v16bf aL0 = load_a_from_f32(l0, kbase);
      v16bf aR0 = load_a_from_f32(r0, kbase);
      v16bf aL1 = load_a_from_f32(l1, kbase);
      v16bf aR1 = load_a_from_f32(r1, kbase);
#pragma unroll
      for (int g = 0; g < 5; ++g) {
        v16bf bl = load_ab_bf16(wrow[g], kbase);       // WL[g] rows
        v16bf br = load_ab_bf16(wrow[5 + g], kbase);   // WR[g] rows
        acc[g][0] = wmma_bf16(aL0, bl, acc[g][0]);
        acc[g][0] = wmma_bf16(aR0, br, acc[g][0]);
        acc[g][1] = wmma_bf16(aL1, bl, acc[g][1]);
        acc[g][1] = wmma_bf16(aR1, br, acc[g][1]);
      }
    }

    const int n = nbase + lo;
    float bias[5];
#pragma unroll
    for (int g = 0; g < 5; ++g) bias[g] = bL[g * H + n] + bR[g * H + n];

#pragma unroll
    for (int st = 0; st < 2; ++st) {
#pragma unroll
      for (int v = 0; v < 8; ++v) {
        const int m  = st * 16 + 8 * hi + v;   // C/D: M = v + 8*hi (+16 for st=1)
        int gm = tileBase + m;
        if (gm > N - 1) gm = N - 1;            // clamp for loads; store masked later
        const float* lr = curIn + (size_t)(2 * gm) * H;
        const float lf = lr[n];
        const float rt = lr[H + n];
        const float rl = fsigmoid(acc[0][st][v] + bias[0]);
        const float rr = fsigmoid(acc[1][st][v] + bias[1]);
        const float zl = fsigmoid(acc[2][st][v] + bias[2]);
        const float zr = fsigmoid(acc[3][st][v] + bias[3]);
        const float zz = fsigmoid(acc[4][st][v] + bias[4]);
        part[jj][st][v] = zl * lf + zr * rt;
        zzs [jj][st][v] = zz;
        uS[m * H + n] = (bf16_t)(rl * lf);
        vS[m * H + n] = (bf16_t)(rr * rt);
      }
    }
  }
  __syncthreads();

  // ---------------- Phase 2: candidate GEMM + combine + store -------------
#pragma unroll
  for (int jj = 0; jj < 2; ++jj) {
    const int nbase = (wave + jj * 8) * 16;
    const bf16_t* wlhRow = wb + (size_t)10 * W_ELEMS + (size_t)(nbase + lo) * H;
    const bf16_t* wrhRow = wb + (size_t)11 * W_ELEMS + (size_t)(nbase + lo) * H;
    const bf16_t* u0 = &uS[(size_t)lo * H];
    const bf16_t* v0 = &vS[(size_t)lo * H];
    const bf16_t* u1 = &uS[(size_t)(16 + lo) * H];
    const bf16_t* v1 = &vS[(size_t)(16 + lo) * H];

    v8f acc[2];
#pragma unroll
    for (int st = 0; st < 2; ++st)
      acc[st] = (v8f){0.f, 0.f, 0.f, 0.f, 0.f, 0.f, 0.f, 0.f};

#pragma unroll
    for (int kc = 0; kc < 8; ++kc) {
      const int kbase = kc * 32 + hi * 8;
      v16bf bl = load_ab_bf16(wlhRow, kbase);
      v16bf br = load_ab_bf16(wrhRow, kbase);
      acc[0] = wmma_bf16(load_ab_bf16(u0, kbase), bl, acc[0]);
      acc[0] = wmma_bf16(load_ab_bf16(v0, kbase), br, acc[0]);
      acc[1] = wmma_bf16(load_ab_bf16(u1, kbase), bl, acc[1]);
      acc[1] = wmma_bf16(load_ab_bf16(v1, kbase), br, acc[1]);
    }

    const int n = nbase + lo;
    const float hb = blh[n] + brh[n];
#pragma unroll
    for (int st = 0; st < 2; ++st) {
#pragma unroll
      for (int v = 0; v < 8; ++v) {
        const int m  = st * 16 + 8 * hi + v;
        const int gm = tileBase + m;
        const float ht = tanhf(acc[st][v] + hb);
        if (gm < N) {
          curOut[(size_t)gm * H + n] = part[jj][st][v] + zzs[jj][st][v] * ht;
        }
      }
    }
  }
}

// ---------------------------------------------------------------------------
// Kernel 4: replicate final 256-vector h into the 3-tuple output (768 floats)
// ---------------------------------------------------------------------------
__global__ void finalize_kernel(const float* __restrict__ h,
                                float* __restrict__ out, int total) {
  int i = blockIdx.x * blockDim.x + threadIdx.x;
  if (i < total) out[i] = h[i & (H - 1)];
}

// ---------------------------------------------------------------------------
// Host-side orchestration (graph-capture safe: only kernel launches).
// Inputs: tokens, emb, WL, WR, bL, bR, Wlh, Wrh, blh, brh
// ---------------------------------------------------------------------------
extern "C" void kernel_launch(void* const* d_in, const int* in_sizes, int n_in,
                              void* d_out, int out_size, void* d_ws, size_t ws_size,
                              hipStream_t stream) {
  (void)in_sizes; (void)n_in; (void)ws_size;

  const int*   tokens = (const int*)  d_in[0];
  const float* emb    = (const float*)d_in[1];
  const float* WL     = (const float*)d_in[2];
  const float* WR     = (const float*)d_in[3];
  const float* bL     = (const float*)d_in[4];
  const float* bR     = (const float*)d_in[5];
  const float* Wlh    = (const float*)d_in[6];
  const float* Wrh    = (const float*)d_in[7];
  const float* blh    = (const float*)d_in[8];
  const float* brh    = (const float*)d_in[9];

  char* ws = (char*)d_ws;
  bf16_t* wb   = reinterpret_cast<bf16_t*>(ws);                               // 1.5 MB
  float*  bufA = reinterpret_cast<float*>(ws + (4ull << 20));                 // 64 MB
  float*  bufB = reinterpret_cast<float*>(ws + (4ull << 20) + (64ull << 20)); // 32 MB

  // 1) Pack weights to bf16 once (hot in L2 across all 16 levels).
  {
    const int total = 12 * W_ELEMS;
    cvt_weights_kernel<<<(total + 255) / 256, 256, 0, stream>>>(WL, WR, Wlh, Wrh, wb);
  }

  // 2) Embedding gather into bufA (level-0 input, 65536 x 256 f32).
  gather_kernel<<<L_TOKENS, 64, 0, stream>>>(tokens, emb, bufA);

  // 3) 16 tree levels, ping-ponging bufA <-> bufB; stream order = level barrier.
  float* cin  = bufA;
  float* cout = bufB;
  for (int N = L_TOKENS / 2; N >= 1; N >>= 1) {
    const int blocks = (N + M_TILE - 1) / M_TILE;
    tree_level_kernel<<<blocks, 256, 0, stream>>>(cin, cout, wb, bL, bR, blh, brh, N);
    float* t = cin; cin = cout; cout = t;
  }

  // 4) cin now holds the single root row; write (h, h, h).
  finalize_kernel<<<(out_size + 255) / 256, 256, 0, stream>>>(cin, (float*)d_out, out_size);
}
